// ContextualLoss_76020921139519
// MI455X (gfx1250) — matrix operand order repaired and checked by reading, not compiled, not run on hardware
//
#include <hip/hip_runtime.h>
#include <hip/hip_bf16.h>
#include <math.h>

typedef __attribute__((ext_vector_type(16))) __bf16 v16bf;
typedef __attribute__((ext_vector_type(8)))  float  v8f;

#define N_B     4
#define C_CH    256
#define P_PIX   4096
#define NP      (N_B * P_PIX)          // 16384 pixels total
#define BAND_INV 2.0f                  // 1 / band_width (0.5)
#define EPSV    1e-5f
#define LDSA_STRIDE 264                // 256 + 8 bf16 pad (bank-conflict avoidance)
#define SIM_STRIDE  4104               // 4096 + 8 f16 pad; row stride 8208 B (16B aligned)

// ---------------------------------------------------------------- helpers
__device__ inline unsigned short f2bf(float f) {
  unsigned u = __float_as_uint(f);
  u += 0x7FFFu + ((u >> 16) & 1u);     // round-to-nearest-even
  return (unsigned short)(u >> 16);
}

// ---------------------------------------------------------------- K0: per-channel mean of y over (n,h,w)
__global__ void k_ymu(const float* __restrict__ y, float* __restrict__ mu) {
  __shared__ float red[256];
  int c = blockIdx.x, t = threadIdx.x;
  float s = 0.f;
  for (int i = t; i < N_B * P_PIX; i += 256) {
    int n = i >> 12, p = i & 4095;
    s += y[(size_t)n * (C_CH * P_PIX) + (size_t)c * P_PIX + p];
  }
  red[t] = s; __syncthreads();
  for (int off = 128; off > 0; off >>= 1) {
    if (t < off) red[t] += red[t + off];
    __syncthreads();
  }
  if (t == 0) mu[c] = red[0] * (1.0f / (N_B * P_PIX));
}

// ---------------------------------------------------------------- K1: center + L2-normalize, write bf16 pixel-major (N*P, C)
__global__ void k_norm(const float* __restrict__ x, const float* __restrict__ y,
                       const float* __restrict__ mu,
                       unsigned short* __restrict__ xnT, unsigned short* __restrict__ ynT) {
  __shared__ float smu[C_CH];
  int t = threadIdx.x;
  smu[t] = mu[t];
  __syncthreads();
  int wid = t >> 5, lane = t & 31;                  // wave32
  int pixel = (blockIdx.x * 8 + wid) * 32 + lane;   // 64 blocks * 8 waves * 32 lanes = 16384
  int n = pixel >> 12, p = pixel & 4095;
  const float* xb = x + (size_t)n * (C_CH * P_PIX) + p;
  const float* yb = y + (size_t)n * (C_CH * P_PIX) + p;
  float ssx = 0.f, ssy = 0.f;
  for (int c = 0; c < C_CH; ++c) {
    float m  = smu[c];
    float vx = xb[(size_t)c * P_PIX] - m;
    float vy = yb[(size_t)c * P_PIX] - m;
    ssx += vx * vx; ssy += vy * vy;
  }
  float ix = 1.0f / fmaxf(sqrtf(ssx), 1e-12f);
  float iy = 1.0f / fmaxf(sqrtf(ssy), 1e-12f);
  unsigned short* xo = xnT + (size_t)pixel * C_CH;
  unsigned short* yo = ynT + (size_t)pixel * C_CH;
  for (int c0 = 0; c0 < C_CH; c0 += 8) {
    union { unsigned short s[8]; uint4 v; } px, py;
    #pragma unroll
    for (int j = 0; j < 8; ++j) {
      int c = c0 + j; float m = smu[c];
      px.s[j] = f2bf((xb[(size_t)c * P_PIX] - m) * ix);
      py.s[j] = f2bf((yb[(size_t)c * P_PIX] - m) * iy);
    }
    *(uint4*)(xo + c0) = px.v;
    *(uint4*)(yo + c0) = py.v;
  }
}

// ---------------------------------------------------------------- K2: zero the column-max accumulator
__global__ void k_init(unsigned int* __restrict__ colmax) {
  int i = blockIdx.x * blockDim.x + threadIdx.x;
  if (i < NP) colmax[i] = 0u;
}

// ---------------------------------------------------------------- K3: fully fused stripe kernel
// block = 256 thr = 8 waves, owns 32 rows x all 4096 cols of one batch.
// Phase 1: WMMA GEMM (bf16, f32 acc); dist=1-sim parked in 256KB f16 LDS; fused row-max.
// Phase 2: w=exp((1-dist/dmin)/h) in-place in LDS + row-sums.
// Phase 3: per-column max of w/rowsum -> global atomicMax (positive-float-as-uint).
__global__ void __launch_bounds__(256)
k_fused(const unsigned short* __restrict__ xnT,
        const unsigned short* __restrict__ ynT,
        unsigned int* __restrict__ colmax) {
  __shared__ unsigned short ldsA[32 * LDSA_STRIDE];  // 16.9 KB A stage
  __shared__ float rmaxArr[4][32];
  __shared__ float wred[8][32];
  __shared__ float sInvArr[32];
  __shared__ float rsArr[32];
  extern __shared__ _Float16 simLDS[];               // 32 * SIM_STRIDE f16 = 256.5 KB

  const int t = threadIdx.x;
  const int n = blockIdx.y;
  const int rowBase = blockIdx.x * 32;

  // ---- stage A stripe (32 rows x 256 ch bf16) via async global->LDS DMA (ASYNCcnt path)
  {
    int row = t >> 3, chunk = (t & 7) * 32;          // 64 B per thread, coalesced
    const unsigned short* src = xnT + (size_t)(n * P_PIX + rowBase + row) * C_CH + chunk;
    unsigned lo = (unsigned)(size_t)(ldsA + row * LDSA_STRIDE + chunk); // low 32b of LDS aperture = ds offset
    unsigned long long ga = (unsigned long long)(size_t)src;
    asm volatile("global_load_async_to_lds_b128 %0, %1, off offset:0\n\t"
                 "global_load_async_to_lds_b128 %0, %1, off offset:16\n\t"
                 "global_load_async_to_lds_b128 %0, %1, off offset:32\n\t"
                 "global_load_async_to_lds_b128 %0, %1, off offset:48"
                 :: "v"(lo), "v"(ga) : "memory");
    asm volatile("s_wait_asynccnt 0x0" ::: "memory");
  }
  __syncthreads();

  const int wid = t >> 5, lane = t & 31;
  const int wm = wid >> 2, wn = wid & 3;
  const int lh = lane >> 4, l15 = lane & 15;

  // ---- preload this wave's 8 A fragments into registers (ISA 16-bit 16x32 A layout)
  v16bf afrag[8];
  {
    const unsigned short* ldsRow = ldsA + (wm * 16 + l15) * LDSA_STRIDE;
    #pragma unroll
    for (int kk = 0; kk < 8; ++kk) {
      union { uint4 u[2]; v16bf v; } a;
      int k2 = kk * 32 + lh * 8;
      a.u[0] = *(const uint4*)(ldsRow + k2);
      a.u[1] = *(const uint4*)(ldsRow + k2 + 16);
      afrag[kk] = a.v;
    }
  }

  float rmax[8];
  #pragma unroll
  for (int r = 0; r < 8; ++r) rmax[r] = -3.0e38f;

  // ---- Phase 1: GEMM over all column tiles; dist -> f16 LDS, running row max
  for (int qt = 0; qt < P_PIX; qt += 64) {
    int col = qt + wn * 16 + l15;
    const unsigned short* bcol = ynT + (size_t)(n * P_PIX + col) * C_CH;
    __builtin_prefetch(bcol, 0, 1);
    v8f acc = {0.f, 0.f, 0.f, 0.f, 0.f, 0.f, 0.f, 0.f};
    #pragma unroll
    for (int kk = 0; kk < 8; ++kk) {
      union { uint4 u[2]; v16bf v; } b;       // B: lane = column q, K spans 8h..8h+7 and 16+8h..16+8h+7
      const uint4* bp = (const uint4*)(bcol + kk * 32 + lh * 8);
      b.u[0] = bp[0]; b.u[1] = bp[2];
      acc = __builtin_amdgcn_wmma_f32_16x16x32_bf16(false, afrag[kk], false, b.v,
                                                    (short)0, acc, false, false);
    }
    #pragma unroll
    for (int r = 0; r < 8; ++r) {
      int rowL = wm * 16 + lh * 8 + r;        // block-local row 0..31
      float v = acc[r];
      simLDS[rowL * SIM_STRIDE + col] = (_Float16)(1.0f - v);   // store dist
      rmax[r] = fmaxf(rmax[r], v);
    }
  }

  // ---- row max across the 16 lanes sharing each row, then across the 4 wn waves
  #pragma unroll
  for (int r = 0; r < 8; ++r) {
    float v = rmax[r];
    v = fmaxf(v, __shfl_xor(v, 1, 32));
    v = fmaxf(v, __shfl_xor(v, 2, 32));
    v = fmaxf(v, __shfl_xor(v, 4, 32));
    v = fmaxf(v, __shfl_xor(v, 8, 32));
    rmax[r] = v;
  }
  if (l15 == 0) {
    #pragma unroll
    for (int r = 0; r < 8; ++r) rmaxArr[wn][wm * 16 + lh * 8 + r] = rmax[r];
  }
  __syncthreads();
  if (t < 32) {
    float m = fmaxf(fmaxf(rmaxArr[0][t], rmaxArr[1][t]),
                    fmaxf(rmaxArr[2][t], rmaxArr[3][t]));
    sInvArr[t] = 1.0f / ((1.0f - m) + EPSV);  // 1 / (min_q dist + eps)
  }
  __syncthreads();

  // ---- Phase 2: w = exp((1 - dist/dmin)/h), in place; per-row partial sums
  float psum[32];
  #pragma unroll
  for (int r = 0; r < 32; ++r) psum[r] = 0.f;
  for (int q = t; q < P_PIX; q += 256) {
    #pragma unroll
    for (int r = 0; r < 32; ++r) {
      float d = (float)simLDS[r * SIM_STRIDE + q];
      float w = __expf((1.0f - d * sInvArr[r]) * BAND_INV);
      simLDS[r * SIM_STRIDE + q] = (_Float16)w;
      psum[r] += w;
    }
  }
  #pragma unroll
  for (int r = 0; r < 32; ++r) {
    float v = psum[r];
    v += __shfl_xor(v, 1, 32);  v += __shfl_xor(v, 2, 32);
    v += __shfl_xor(v, 4, 32);  v += __shfl_xor(v, 8, 32);
    v += __shfl_xor(v, 16, 32);
    if (lane == 0) wred[wid][r] = v;
  }
  __syncthreads();
  if (t < 32) {
    float s = 0.f;
    #pragma unroll
    for (int w = 0; w < 8; ++w) s += wred[w][t];
    rsArr[t] = 1.0f / s;                       // 1 / rowsum
  }
  __syncthreads();

  // ---- Phase 3: column max of cx = w / rowsum over this block's 32 rows -> global
  for (int q = t; q < P_PIX; q += 256) {
    float m = 0.f;
    #pragma unroll
    for (int r = 0; r < 32; ++r)
      m = fmaxf(m, (float)simLDS[r * SIM_STRIDE + q] * rsArr[r]);
    atomicMax(&colmax[n * P_PIX + q], __float_as_uint(m));  // cx > 0: uint max == float max
  }
}

// ---------------------------------------------------------------- K4: mean over q, -log, mean over n
__global__ void k_loss(const unsigned int* __restrict__ colmax, float* __restrict__ out) {
  __shared__ float red[256];
  int t = threadIdx.x;
  float loss = 0.f;
  for (int n = 0; n < N_B; ++n) {
    float s = 0.f;
    for (int q = t; q < P_PIX; q += 256) s += __uint_as_float(colmax[n * P_PIX + q]);
    red[t] = s; __syncthreads();
    for (int off = 128; off > 0; off >>= 1) {
      if (t < off) red[t] += red[t + off];
      __syncthreads();
    }
    if (t == 0) loss += -logf(red[0] * (1.0f / P_PIX) + EPSV);
    __syncthreads();
  }
  if (t == 0) out[0] = loss * (1.0f / N_B);
}

// ---------------------------------------------------------------- host
extern "C" void kernel_launch(void* const* d_in, const int* in_sizes, int n_in,
                              void* d_out, int out_size, void* d_ws, size_t ws_size,
                              hipStream_t stream) {
  (void)in_sizes; (void)n_in; (void)out_size;
  const float* x = (const float*)d_in[0];
  const float* y = (const float*)d_in[1];

  char* ws = (char*)d_ws;
  size_t o = 0;
  float*          mu  = (float*)(ws + o);           o += 1024;
  unsigned short* xnT = (unsigned short*)(ws + o);  o += (size_t)NP * C_CH * 2;   // 8 MB
  unsigned short* ynT = (unsigned short*)(ws + o);  o += (size_t)NP * C_CH * 2;   // 8 MB
  unsigned int*   cmx = (unsigned int*)(ws + o);    o += (size_t)NP * 4;          // 64 KB
  if (ws_size < o) return;

  const int dynLds = 32 * SIM_STRIDE * (int)sizeof(_Float16);   // 262,656 B
  hipFuncSetAttribute(reinterpret_cast<const void*>(k_fused),
                      hipFuncAttributeMaxDynamicSharedMemorySize, dynLds);

  k_ymu  <<<256, 256, 0, stream>>>(y, mu);
  k_norm <<<64, 256, 0, stream>>>(x, y, mu, xnT, ynT);
  k_init <<<NP / 256, 256, 0, stream>>>(cmx);
  k_fused<<<dim3(P_PIX / 32, N_B), 256, dynLds, stream>>>(xnT, ynT, cmx);
  k_loss <<<1, 256, 0, stream>>>(cmx, (float*)d_out);
}
